// CustomRNNDecoder_60327110640370
// MI455X (gfx1250) — matrix operand order repaired and checked by reading, not compiled, not run on hardware
//
#include <hip/hip_runtime.h>
#include <hip/hip_bf16.h>

// ---------------------------------------------------------------------------
// CDNA5 (gfx1250) GRU decoder.
//   Gi   = embed[tok] @ w_ih^T + b_ih          (parallel GEMM, bf16 WMMA)
//   h_t  = GRU(h_{t-1}, Gi[t])                 (257 sequential step kernels)
//   Out  = H @ lin_w^T + lin_b                 (parallel GEMM, bf16 WMMA)
// Matmuls: v_wmma_f32_16x16x32_bf16 (f32 accumulation).
// LDS staging: 16-byte b128 chunks; async-to-LDS path when available.
// ---------------------------------------------------------------------------

typedef __attribute__((ext_vector_type(16))) __bf16 v16bf;
typedef __attribute__((ext_vector_type(8)))  __bf16 v8bf;
typedef __attribute__((ext_vector_type(4)))  float  v4f;
typedef __attribute__((ext_vector_type(8)))  float  v8f;

#define B_   64
#define D_   512
#define G3_  1536   // 3*D
#define J_   640
#define LDT  40     // padded LDS stride (elems) for 32-wide K tiles; 80B rows

#if defined(__has_builtin)
#  if __has_builtin(__builtin_amdgcn_global_load_async_to_lds_b128)
#    define HAVE_ASYNC_LDS 1
#  endif
#endif
#ifndef HAVE_ASYNC_LDS
#  define HAVE_ASYNC_LDS 0
#endif

#if HAVE_ASYNC_LDS
// Builtin expects: (v4i addrspace(1)*, v4i addrspace(3)*, int off, int cpol)
typedef int v4i_ __attribute__((vector_size(16)));
typedef __attribute__((address_space(1))) v4i_ gv4i_;
typedef __attribute__((address_space(3))) v4i_ lv4i_;
#endif

// Stage one 16-byte chunk global -> LDS.
__device__ __forceinline__ void stage16(const __bf16* g, __bf16* l) {
#if HAVE_ASYNC_LDS
  __builtin_amdgcn_global_load_async_to_lds_b128((gv4i_*)g, (lv4i_*)l, 0, 0);
#else
  *(v8bf*)l = *(const v8bf*)g;
#endif
}

// Make staged data visible to the workgroup.
__device__ __forceinline__ void stage_commit() {
#if HAVE_ASYNC_LDS
#  if __has_builtin(__builtin_amdgcn_s_wait_asynccnt)
  __builtin_amdgcn_s_wait_asynccnt(0);
#  else
  asm volatile("s_wait_asynccnt 0" ::: "memory");
#  endif
#endif
  __syncthreads();
}

__device__ __forceinline__ v8f wmma_bf16(v16bf a, v16bf b, v8f c) {
  return __builtin_amdgcn_wmma_f32_16x16x32_bf16(
      /*neg_a=*/false, a, /*neg_b=*/false, b,
      /*c_mod=*/(short)0, c, /*reuse_a=*/false, /*reuse_b=*/false);
}

// A fragment (16x32 bf16, MxK): lanes 0-15 row M=lane, K 0..15;
// lanes 16-31 row M=lane-16, K 16..31. Two 16B LDS reads per lane.
__device__ __forceinline__ v16bf load_a_frag(const __bf16* Abase, int ld) {
  const int lane = threadIdx.x & 31;
  const int row  = lane & 15;
  const int kh   = (lane >> 4) << 4;   // 0 or 16
  const __bf16* p = Abase + row * ld + kh;
  v16bf a;
#pragma unroll
  for (int i = 0; i < 8; ++i) a[i] = p[i];        // K = kh + 0..7
  p += 8;
#pragma unroll
  for (int i = 0; i < 8; ++i) a[8 + i] = p[i];    // K = kh + 8..15
  return a;
}

// B fragment (32x16 bf16, KxN) from row-major weight tile W[n][k]
// (B[k][n] = W[n][k]): lanes 0-15 col N=lane K 0..15; lanes 16-31 col
// N=lane-16 K 16..31. Two 16B LDS reads per lane.
__device__ __forceinline__ v16bf load_b_frag(const __bf16* Wbase, int ld) {
  const int lane = threadIdx.x & 31;
  const int col  = lane & 15;
  const int kofs = (lane >> 4) << 4;   // 0 or 16
  const __bf16* p = Wbase + col * ld + kofs;
  v16bf b;
#pragma unroll
  for (int i = 0; i < 16; ++i) b[i] = p[i];
  return b;
}

__device__ __forceinline__ float sigmoidf_(float x) {
  return 1.0f / (1.0f + __expf(-x));
}

// ---------------------------------------------------------------------------
// fp32 -> bf16 conversion, 4 elems/thread (b128 in, b64 out)
// ---------------------------------------------------------------------------
__global__ void cvt_bf16_kernel(const float* __restrict__ in,
                                __bf16* __restrict__ out, size_t n4) {
  size_t i = (size_t)blockIdx.x * blockDim.x + threadIdx.x;
  size_t stride = (size_t)gridDim.x * blockDim.x;
  for (; i < n4; i += stride) {
    v4f v = *(const v4f*)(in + 4 * i);
    __bf16 tmp[4] = {(__bf16)v.x, (__bf16)v.y, (__bf16)v.z, (__bf16)v.w};
    *(uint64_t*)(out + 4 * i) = *(const uint64_t*)tmp;
  }
}

// h0 = broadcast(init_state) -> Hf slot 0 (f32) and Hb slot 0 (bf16)
__global__ void init_h_kernel(const float* __restrict__ init_state,
                              float* __restrict__ Hf, __bf16* __restrict__ Hb) {
  int i = blockIdx.x * blockDim.x + threadIdx.x;   // 0 .. B*D-1
  float v = init_state[i & (D_ - 1)];
  Hf[i] = v;
  Hb[i] = (__bf16)v;
}

// ---------------------------------------------------------------------------
// Gi[t*64+b, j] = embed[tok(t,b)] . w_ih[j] + b_ih[j]
// grid = (U+1, 24), block = 256 (8 waves). Block tile 64x64, K tiled by 32.
// ---------------------------------------------------------------------------
__global__ void gi_gemm_kernel(const int* __restrict__ src_tokens,
                               const __bf16* __restrict__ emb,
                               const __bf16* __restrict__ wih,
                               const float* __restrict__ b_ih,
                               float* __restrict__ Gi, int U) {
  const int t    = blockIdx.x;        // 0..U   (one timestep == 64 rows)
  const int nblk = blockIdx.y;        // 0..23
  const int tid  = threadIdx.x;

  __shared__ __bf16 As[B_ * LDT];
  __shared__ __bf16 Ws[B_ * LDT];

  // Staging assignment: thread -> one 16B chunk of each 64x32 tile.
  const int r    = tid >> 2;          // row 0..63
  const int cofs = (tid & 3) << 3;    // 0,8,16,24
  const int token = (t == 0) ? 0 : src_tokens[(size_t)r * U + (t - 1)];
  const __bf16* gA = emb + (size_t)token * D_ + cofs;
  const __bf16* gW = wih + (size_t)((nblk << 6) + r) * D_ + cofs;
  __bf16* lA = &As[r * LDT + cofs];
  __bf16* lW = &Ws[r * LDT + cofs];

  const int wave = tid >> 5, lane = tid & 31;
  const int msub  = (wave >> 1) << 4;      // 0,16,32,48
  const int nbase = (wave & 1) << 5;       // 0,32

  v8f c0 = {}; v8f c1 = {};

  for (int k0 = 0; k0 < D_; k0 += 32) {
    __syncthreads();
    stage16(gA + k0, lA);
    stage16(gW + k0, lW);
    if (k0 + 32 < D_) __builtin_prefetch(gW + k0 + 32, 0, 1);
    stage_commit();
    v16bf a  = load_a_frag(&As[msub * LDT], LDT);
    v16bf b0 = load_b_frag(&Ws[nbase * LDT], LDT);
    v16bf b1 = load_b_frag(&Ws[(nbase + 16) * LDT], LDT);
    c0 = wmma_bf16(a, b0, c0);
    c1 = wmma_bf16(a, b1, c1);
  }

#pragma unroll
  for (int i = 0; i < 8; ++i) {
    int m  = msub + i + ((lane >> 4) << 3);       // batch index within 64
    int n0 = (nblk << 6) + nbase + (lane & 15);
    size_t base = ((size_t)(t * B_ + m)) * G3_;
    Gi[base + n0]      = c0[i] + b_ih[n0];
    Gi[base + n0 + 16] = c1[i] + b_ih[n0 + 16];
  }
}

// ---------------------------------------------------------------------------
// One GRU step: gh = h_{t-1} @ w_hh^T (three 64x64 gate tiles per block),
// gates, h_t. grid = 8 blocks (d-chunks of 64), block = 256.
// Hb/Hf slot layout: slot 0 = h0, slot t+1 = h_t.
// ---------------------------------------------------------------------------
__global__ void gru_step_kernel(const float* __restrict__ Gi,
                                const __bf16* __restrict__ whh,
                                const float* __restrict__ b_hh,
                                float* __restrict__ Hf,
                                __bf16* __restrict__ Hb, int t) {
  const int dblk  = blockIdx.x;        // 0..7
  const int tid   = threadIdx.x;
  const int dbase = dblk << 6;

  __shared__ __bf16 As[B_ * LDT];
  __shared__ __bf16 Ws[3][B_ * LDT];

  const __bf16* hb  = Hb + (size_t)t * (B_ * D_);
  const float*  hf  = Hf + (size_t)t * (B_ * D_);
  float*  hf_out    = Hf + (size_t)(t + 1) * (B_ * D_);
  __bf16* hb_out    = Hb + (size_t)(t + 1) * (B_ * D_);

  const int r    = tid >> 2;
  const int cofs = (tid & 3) << 3;
  const __bf16* gA  = hb + (size_t)r * D_ + cofs;
  const __bf16* gW0 = whh + (size_t)(0 * D_ + dbase + r) * D_ + cofs;
  const __bf16* gW1 = whh + (size_t)(1 * D_ + dbase + r) * D_ + cofs;
  const __bf16* gW2 = whh + (size_t)(2 * D_ + dbase + r) * D_ + cofs;
  __bf16* lA  = &As[r * LDT + cofs];
  __bf16* lW0 = &Ws[0][r * LDT + cofs];
  __bf16* lW1 = &Ws[1][r * LDT + cofs];
  __bf16* lW2 = &Ws[2][r * LDT + cofs];

  const int wave = tid >> 5, lane = tid & 31;
  const int msub  = (wave >> 1) << 4;
  const int nbase = (wave & 1) << 5;

  v8f acc[3][2] = {};

  for (int k0 = 0; k0 < D_; k0 += 32) {
    __syncthreads();
    stage16(gA  + k0, lA);
    stage16(gW0 + k0, lW0);
    stage16(gW1 + k0, lW1);
    stage16(gW2 + k0, lW2);
    stage_commit();
    v16bf a = load_a_frag(&As[msub * LDT], LDT);
#pragma unroll
    for (int g = 0; g < 3; ++g) {
      v16bf b0 = load_b_frag(&Ws[g][nbase * LDT], LDT);
      v16bf b1 = load_b_frag(&Ws[g][(nbase + 16) * LDT], LDT);
      acc[g][0] = wmma_bf16(a, b0, acc[g][0]);
      acc[g][1] = wmma_bf16(a, b1, acc[g][1]);
    }
  }

#pragma unroll
  for (int i = 0; i < 8; ++i) {
#pragma unroll
    for (int half = 0; half < 2; ++half) {
      int m = msub + i + ((lane >> 4) << 3);              // batch b
      int d = dbase + nbase + (half << 4) + (lane & 15);  // feature d
      float ghr = acc[0][half][i] + b_hh[d];
      float ghz = acc[1][half][i] + b_hh[D_ + d];
      float ghn = acc[2][half][i] + b_hh[2 * D_ + d];
      size_t gbase = ((size_t)(t * B_ + m)) * G3_;
      float ir  = Gi[gbase + d];
      float iz  = Gi[gbase + D_ + d];
      float in_ = Gi[gbase + 2 * D_ + d];
      float rr = sigmoidf_(ir + ghr);
      float zz = sigmoidf_(iz + ghz);
      float nn = tanhf(in_ + rr * ghn);
      float hp = hf[m * D_ + d];
      float hv = (1.0f - zz) * nn + zz * hp;
      hf_out[m * D_ + d] = hv;
      hb_out[m * D_ + d] = (__bf16)hv;
    }
  }
}

// ---------------------------------------------------------------------------
// Out[b, t, j] = H_t[b] . lin_w[j] + lin_b[j]   (transposed write)
// grid = (U+1, 10), block = 256. Block tile 64 rows x 64 cols, K tiled by 32.
// ---------------------------------------------------------------------------
__global__ void out_gemm_kernel(const __bf16* __restrict__ Hb,
                                const __bf16* __restrict__ linw,
                                const float* __restrict__ lin_b,
                                float* __restrict__ out, int U) {
  const int t    = blockIdx.x;        // 0..U
  const int nblk = blockIdx.y;        // 0..9
  const int tid  = threadIdx.x;

  __shared__ __bf16 As[B_ * LDT];
  __shared__ __bf16 Ws[B_ * LDT];

  const __bf16* hbase = Hb + (size_t)(t + 1) * (B_ * D_);

  const int r    = tid >> 2;
  const int cofs = (tid & 3) << 3;
  const __bf16* gA = hbase + (size_t)r * D_ + cofs;
  const __bf16* gW = linw + (size_t)((nblk << 6) + r) * D_ + cofs;
  __bf16* lA = &As[r * LDT + cofs];
  __bf16* lW = &Ws[r * LDT + cofs];

  const int wave = tid >> 5, lane = tid & 31;
  const int msub  = (wave >> 1) << 4;
  const int nbase = (wave & 1) << 5;

  v8f c0 = {}; v8f c1 = {};

  for (int k0 = 0; k0 < D_; k0 += 32) {
    __syncthreads();
    stage16(gA + k0, lA);
    stage16(gW + k0, lW);
    if (k0 + 32 < D_) __builtin_prefetch(gW + k0 + 32, 0, 1);
    stage_commit();
    v16bf a  = load_a_frag(&As[msub * LDT], LDT);
    v16bf b0 = load_b_frag(&Ws[nbase * LDT], LDT);
    v16bf b1 = load_b_frag(&Ws[(nbase + 16) * LDT], LDT);
    c0 = wmma_bf16(a, b0, c0);
    c1 = wmma_bf16(a, b1, c1);
  }

  const int Up1 = U + 1;
#pragma unroll
  for (int i = 0; i < 8; ++i) {
    int m  = msub + i + ((lane >> 4) << 3);     // batch b
    int j0 = (nblk << 6) + nbase + (lane & 15);
    size_t base = ((size_t)m * Up1 + t) * J_;
    out[base + j0]      = c0[i] + lin_b[j0];
    out[base + j0 + 16] = c1[i] + lin_b[j0 + 16];
  }
}

// ---------------------------------------------------------------------------
// Host-side launch
// ---------------------------------------------------------------------------
extern "C" void kernel_launch(void* const* d_in, const int* in_sizes, int n_in,
                              void* d_out, int out_size, void* d_ws, size_t ws_size,
                              hipStream_t stream) {
  const int*   src_tokens = (const int*)  d_in[0];
  // d_in[1] = src_lengths (unused by the reference)
  const float* embed_w    = (const float*)d_in[2];
  const float* w_ih       = (const float*)d_in[3];
  const float* w_hh       = (const float*)d_in[4];
  const float* b_ih       = (const float*)d_in[5];
  const float* b_hh       = (const float*)d_in[6];
  const float* lin_w      = (const float*)d_in[7];
  const float* lin_b      = (const float*)d_in[8];
  const float* init_state = (const float*)d_in[9];
  float* out = (float*)d_out;

  const int U = in_sizes[0] / B_;          // 256
  const int V = in_sizes[2] / D_;          // 32000
  const int Up1 = U + 1;

  // Workspace carve-up (256B aligned)
  char* p = (char*)d_ws;
  auto carve = [&](size_t bytes) -> char* {
    char* r = p;
    p += (bytes + 255) & ~(size_t)255;
    return r;
  };
  __bf16* emb_bf = (__bf16*)carve((size_t)V * D_ * sizeof(__bf16));
  __bf16* wih_bf = (__bf16*)carve((size_t)G3_ * D_ * sizeof(__bf16));
  __bf16* whh_bf = (__bf16*)carve((size_t)G3_ * D_ * sizeof(__bf16));
  __bf16* lin_bf = (__bf16*)carve((size_t)J_ * D_ * sizeof(__bf16));
  float*  Gi     = (float*) carve((size_t)Up1 * B_ * G3_ * sizeof(float));
  float*  Hf     = (float*) carve((size_t)(Up1 + 1) * B_ * D_ * sizeof(float));
  __bf16* Hb     = (__bf16*)carve((size_t)(Up1 + 1) * B_ * D_ * sizeof(__bf16));

  // 1) weight / embedding conversion to bf16 (4 elems/thread)
  cvt_bf16_kernel<<<2048, 256, 0, stream>>>(embed_w, emb_bf, (size_t)V * D_ / 4);
  cvt_bf16_kernel<<<512,  256, 0, stream>>>(w_ih,  wih_bf, (size_t)G3_ * D_ / 4);
  cvt_bf16_kernel<<<512,  256, 0, stream>>>(w_hh,  whh_bf, (size_t)G3_ * D_ / 4);
  cvt_bf16_kernel<<<256,  256, 0, stream>>>(lin_w, lin_bf, (size_t)J_ * D_ / 4);

  // 2) h0 broadcast
  init_h_kernel<<<(B_ * D_) / 256, 256, 0, stream>>>(init_state, Hf, Hb);

  // 3) input projection for all timesteps (parallel)
  gi_gemm_kernel<<<dim3(Up1, G3_ / 64), 256, 0, stream>>>(
      src_tokens, emb_bf, wih_bf, b_ih, Gi, U);

  // 4) sequential GRU recurrence (graph-captured launch per step gives the
  //    required grid-wide dependency between steps)
  for (int t = 0; t < Up1; ++t)
    gru_step_kernel<<<D_ / 64, 256, 0, stream>>>(Gi, whh_bf, b_hh, Hf, Hb, t);

  // 5) output projection for all timesteps (parallel), transposed write
  out_gemm_kernel<<<dim3(Up1, J_ / 64), 256, 0, stream>>>(
      Hb, lin_bf, lin_b, out, U);
}
